// MLog_83210696393294
// MI455X (gfx1250) — matrix operand order repaired and checked by reading, not compiled, not run on hardware
//
#include <hip/hip_runtime.h>
#include <hip/hip_fp16.h>

// ---------------------------------------------------------------------------
// MI455X (gfx1250) implementation.
//
// Shapes: B=32, S=2048, D_IN=300, D2=300 (pad 384), H=256, gates 4H=1024.
// Roofline: ~107 GFLOP total, ~80 MB traffic -> latency-bound on the 2048-step
// recurrence, not HBM. Strategy: persistent 8-workgroup LSTM, all recurrent
// weights resident in LDS (284.5KB/WGP of the 320KB budget) as f16 in WMMA-B
// (transposed) layout, per-step GEMMs via v_wmma_f32_16x16x32_f16 (wave32),
// fp32 accumulate + fp32 cell state, 3 L2 atomic barriers per step.
// GLOBAL_LOAD_ASYNC_TO_LDS_B128 for all global->LDS copies (ASYNCcnt-tracked,
// bypasses VGPRs) and global_prefetch for next-step X.
// ---------------------------------------------------------------------------

typedef _Float16 half_t;
typedef _Float16 v16h __attribute__((ext_vector_type(16)));
typedef _Float16 v8h  __attribute__((ext_vector_type(8)));
typedef float    v8f  __attribute__((ext_vector_type(8)));
typedef int      v4i_t __attribute__((ext_vector_type(4)));

#define NWG 8      // persistent workgroups in LSTM kernel
#define D2P 384    // D2 300 padded (8 slices of 48 = 3 N-tiles)
#define KINP 320   // D_IN 300 padded for the input projection

// ---- async global->LDS copy (CDNA5 GLOBAL_LOAD_ASYNC_TO_LDS_B128) ----------
#if defined(__AMDGCN__) &&                                            \
    __has_builtin(__builtin_amdgcn_global_load_async_to_lds_b128) &&  \
    __has_builtin(__builtin_amdgcn_s_wait_asynccnt)
#define ASYNC_G2S 1
#else
#define ASYNC_G2S 0
#endif

__device__ __forceinline__ void cp16(void* lds, const void* g) {
#if ASYNC_G2S
  // builtin signature (from hipcc diagnostic): (v4i AS1*, v4i AS3*, imm, imm)
  __builtin_amdgcn_global_load_async_to_lds_b128(
      (__attribute__((address_space(1))) v4i_t*)(void*)g,
      (__attribute__((address_space(3))) v4i_t*)lds, 0, 0);
#else
  *(int4*)lds = *(const int4*)g;
#endif
}
__device__ __forceinline__ void cp_fence() {
#if ASYNC_G2S
  __builtin_amdgcn_s_wait_asynccnt(0);  // all async LDS writes visible
#endif
}

__device__ __forceinline__ v8f wmma_f16(v16h a, v16h b, v8f c) {
  // D = A(16x32) * B(32x16) + C, f32 accumulate
  return __builtin_amdgcn_wmma_f32_16x16x32_f16(false, a, false, b, (short)0, c,
                                                false, false);
}

// A fragment (16x32 f16, row-major source, stride ld):
// lanes 0-15: row M=lane, halves[0..7]=K k0..k0+7, halves[8..15]=K k0+16..23
// lanes 16-31: row M=lane-16, K offsets +8 / +24.
__device__ __forceinline__ v16h ld_a(const half_t* base, int ld, int m0, int k0,
                                     int lane) {
  int lr = lane & 15;
  int kh = (lane >> 4) << 3;
  const half_t* p = base + (m0 + lr) * ld + k0 + kh;
  union { v16h v; v8h h[2]; } u;
  u.h[0] = *(const v8h*)p;
  u.h[1] = *(const v8h*)(p + 16);
  return u.v;
}

// B fragment (32x16 f16) from N-major (transposed) weights, stride ld:
// lane holds column N=lane&15; lanes 0-15 carry K k0..k0+7 and k0+16..23, lanes 16-31 K+8 / +24.
__device__ __forceinline__ v16h ld_b(const half_t* base, int ld, int n0, int k0,
                                     int lane) {
  int lr = lane & 15;
  int kh = (lane >> 4) << 3;
  const half_t* p = base + (n0 + lr) * ld + k0 + kh;
  union { v16h v; v8h h[2]; } u;
  u.h[0] = *(const v8h*)p;
  u.h[1] = *(const v8h*)(p + 16);
  return u.v;
}

// D/C layout: VGPR r -> row m0 + (lane>>4)*8 + r, col n0 + (lane&15)
__device__ __forceinline__ void st_d(float* out, int ld, int m0, int n0, int lane,
                                     v8f acc) {
  int lr = lane & 15;
  int mh = (lane >> 4) << 3;
#pragma unroll
  for (int r = 0; r < 8; ++r) out[(m0 + mh + r) * ld + n0 + lr] = acc[r];
}

__device__ __forceinline__ float sigf(float x) { return 1.f / (1.f + __expf(-x)); }
// NaN-free tanh: 1 - 2/(e^{2x}+1)
__device__ __forceinline__ float tanh_f(float x) {
  float e = __expf(2.f * x);
  return 1.f - 2.f / (e + 1.f);
}

// monotonic-counter grid barrier across the 8 persistent workgroups (L2 spin)
__device__ __forceinline__ void gbar(unsigned* cnt, unsigned target) {
  __syncthreads();
  if (threadIdx.x == 0) {
    __threadfence();
    atomicAdd(cnt, 1u);
    while (__hip_atomic_load(cnt, __ATOMIC_ACQUIRE, __HIP_MEMORY_SCOPE_AGENT) <
           target)
      __builtin_amdgcn_s_sleep(2);
  }
  __syncthreads();
}

// ---------------------------------------------------------------------------
// Prep: fp32 (Ksrc x Nsrc) row-major -> f16 transposed+padded (Npad x Kpad)
// so each WMMA-B lane can read its column's K values contiguously.
// ---------------------------------------------------------------------------
__global__ void MLog_prep_kernel(const float* __restrict__ src,
                                 half_t* __restrict__ dst, int Ksrc, int Nsrc,
                                 int Kpad, int Npad) {
  int total = Kpad * Npad;
  for (int id = blockIdx.x * blockDim.x + threadIdx.x; id < total;
       id += gridDim.x * blockDim.x) {
    int n = id / Kpad, k = id % Kpad;
    dst[id] = (n < Nsrc && k < Ksrc) ? (half_t)src[k * Nsrc + n] : (half_t)0.f;
  }
}

// ---------------------------------------------------------------------------
// Kernel 1: Xh[t][b][0..383] = tanh(features[b][t][:] @ W_in + b_in)   (f16)
// 65536x300x300 GEMM with W_in^T cached in LDS (245KB), WMMA f16.
// ---------------------------------------------------------------------------
__device__ __forceinline__ v16h ld_a_f32(const float* rowp, int k0, int lane,
                                         int kmax) {
  int kh = (lane >> 4) << 3;
  union { v16h v; half_t e[16]; } u;
#pragma unroll
  for (int i = 0; i < 8; ++i) {
    int ka = k0 + kh + i, kb = k0 + 16 + kh + i;
    u.e[i]     = (ka < kmax) ? (half_t)rowp[ka] : (half_t)0.f;
    u.e[8 + i] = (kb < kmax) ? (half_t)rowp[kb] : (half_t)0.f;
  }
  return u.v;
}

__global__ __launch_bounds__(256) void MLog_inproj_kernel(
    const float* __restrict__ features, const float* __restrict__ b_in,
    const half_t* __restrict__ WinT, half_t* __restrict__ Xh) {
  extern __shared__ char smem[];
  half_t* wlds = (half_t*)smem;  // [384][320] f16, N-major
  for (int i = threadIdx.x; i < (384 * KINP) / 8; i += 256)
    cp16((char*)wlds + i * 16, (const char*)WinT + i * 16);
  cp_fence();
  __syncthreads();
  int wave = threadIdx.x >> 5, lane = threadIdx.x & 31;
  int lr = lane & 15, mh = (lane >> 4) << 3;
  for (int it = 0; it < 8; ++it) {
    int rt = blockIdx.x * 8 + it;  // 4096 row tiles: (t, b-half)
    int t = rt >> 1, b0 = (rt & 1) * 16;
    const float* rowp = features + ((size_t)(b0 + lr) * 2048 + t) * 300;
    v8f acc0 = {}, acc1 = {}, acc2 = {};
    for (int k0 = 0; k0 < KINP; k0 += 32) {
      v16h a = ld_a_f32(rowp, k0, lane, 300);
      acc0 = wmma_f16(a, ld_b(wlds, KINP, (wave * 3 + 0) * 16, k0, lane), acc0);
      acc1 = wmma_f16(a, ld_b(wlds, KINP, (wave * 3 + 1) * 16, k0, lane), acc1);
      acc2 = wmma_f16(a, ld_b(wlds, KINP, (wave * 3 + 2) * 16, k0, lane), acc2);
    }
#pragma unroll
    for (int j = 0; j < 3; ++j) {
      v8f acc = (j == 0) ? acc0 : ((j == 1) ? acc1 : acc2);
      int d = (wave * 3 + j) * 16 + lr;
      float bb = (d < 300) ? b_in[d] : 0.f;
#pragma unroll
      for (int r = 0; r < 8; ++r) {
        int b = b0 + mh + r;
        Xh[((size_t)t * 32 + b) * D2P + d] = (half_t)tanh_f(acc[r] + bb);
      }
    }
  }
}

// ---------------------------------------------------------------------------
// Kernel 2: persistent Mogrifier-LSTM. 8 WGs x 256 threads. WG g owns
// h-cols [g*32, g*32+32), gate cols q*256+g*32+i, d2-slice [g*48, g*48+48).
// Per step: P1 sx=h@Q(slice) -> publish x_t; P2 sh=x@R(slice) -> publish h_mog;
// P3 gates = x@Wih(slice)+h_mog@Whh(slice) -> cell update -> publish h.
// ---------------------------------------------------------------------------
__global__ __launch_bounds__(256) void MLog_lstm_kernel(
    const half_t* __restrict__ Xh, const half_t* __restrict__ QtG,
    const half_t* __restrict__ RtG, const half_t* __restrict__ WihTG,
    const half_t* __restrict__ WhhTG, const float* __restrict__ bih,
    const float* __restrict__ bhh, half_t* __restrict__ xsh,
    half_t* __restrict__ hmsh, half_t* __restrict__ hsh,
    half_t* __restrict__ hsOut, unsigned* __restrict__ syncCnt) {
  extern __shared__ char smem[];
  half_t* qt  = (half_t*)(smem);            // [48][256]
  half_t* rt  = (half_t*)(smem + 24576);    // [32][384]
  half_t* wih = (half_t*)(smem + 49152);    // [128][384]
  half_t* whh = (half_t*)(smem + 147456);   // [128][256]
  half_t* xL  = (half_t*)(smem + 212992);   // [32][384] full x_t
  half_t* hL  = (half_t*)(smem + 237568);   // [32][256] full h
  half_t* hmL = (half_t*)(smem + 253952);   // [32][256] full h_mog
  float*  scr = (float*)(smem + 270336);    // [32][128] GEMM staging (reused)
  float*  cL  = (float*)(smem + 286720);    // [32][32] cell state slice
  float*  bL  = (float*)(smem + 290816);    // [128] combined gate bias slice

  const int g = blockIdx.x, j0 = g * 32, d0 = g * 48;
  const int tid = threadIdx.x, wave = tid >> 5, lane = tid & 31;
  const int4 zero4 = {0, 0, 0, 0};

  // ---- one-time weight staging into LDS (async DMA; resident 2048 steps) ----
  for (int c = tid; c < 48 * 32; c += 256) {  // Qt slice: rows d0..d0+47
    int r = c >> 5, off = (c & 31) << 3;
    cp16(qt + r * 256 + off, QtG + (size_t)(d0 + r) * 256 + off);
  }
  for (int c = tid; c < 32 * 48; c += 256) {  // Rt slice: rows j0..j0+31
    int r = c / 48, off = (c % 48) << 3;
    cp16(rt + r * D2P + off, RtG + (size_t)(j0 + r) * D2P + off);
  }
  for (int c = tid; c < 128 * 48; c += 256) {  // WihT slice (gate cols)
    int r = c / 48, off = (c % 48) << 3;
    int gr = ((r >> 5) << 8) + j0 + (r & 31);
    cp16(wih + r * D2P + off, WihTG + (size_t)gr * D2P + off);
  }
  for (int c = tid; c < 128 * 32; c += 256) {  // WhhT slice
    int r = c >> 5, off = (c & 31) << 3;
    int gr = ((r >> 5) << 8) + j0 + (r & 31);
    cp16(whh + r * 256 + off, WhhTG + (size_t)gr * 256 + off);
  }
  for (int i = tid; i < 32 * 256 / 8; i += 256) ((int4*)hL)[i] = zero4;  // h0=0
  for (int i = tid; i < 32 * 32 / 4; i += 256) ((int4*)cL)[i] = zero4;   // c0=0
  if (tid < 128) {
    int q = tid >> 5, i = tid & 31, col = (q << 8) + j0 + i;
    bL[tid] = bih[col] + bhh[col];
  }
  cp_fence();
  __syncthreads();

  unsigned gen = 0;
  for (int t = 0; t < 2048; ++t) {
    // prefetch next step's X slice (rows b=tid, cols d0..d0+47) into caches
    if (tid < 32 && t + 1 < 2048)
      __builtin_prefetch(Xh + ((size_t)(t + 1) * 32 + tid) * D2P + d0, 0, 3);

    // ---- phase 1: sx = h(32x256) @ Qt-slice(256x48) ----
    if (wave < 6) {
      int mi = wave & 1, ni = wave >> 1;
      v8f acc = {};
      for (int k = 0; k < 256; k += 32)
        acc = wmma_f16(ld_a(hL, 256, mi * 16, k, lane),
                       ld_b(qt, 256, ni * 16, k, lane), acc);
      st_d(scr, 48, mi * 16, ni * 16, lane, acc);
    }
    __syncthreads();
    for (int idx = tid; idx < 32 * 48; idx += 256) {
      int b = idx / 48, d = idx % 48;
      float xr = (float)Xh[((size_t)t * 32 + b) * D2P + d0 + d];
      xsh[b * D2P + d0 + d] = (half_t)(2.f * sigf(scr[b * 48 + d]) * xr);
    }
    gen += NWG; gbar(syncCnt, gen);
    for (int i = tid; i < 32 * D2P / 8; i += 256)
      cp16((char*)xL + i * 16, (const char*)xsh + i * 16);
    cp_fence();
    __syncthreads();

    // ---- phase 2: sh = x(32x384) @ Rt-slice(384x32) ----
    if (wave < 4) {
      int mi = wave & 1, ni = wave >> 1;
      v8f acc = {};
      for (int k = 0; k < D2P; k += 32)
        acc = wmma_f16(ld_a(xL, D2P, mi * 16, k, lane),
                       ld_b(rt, D2P, ni * 16, k, lane), acc);
      st_d(scr, 32, mi * 16, ni * 16, lane, acc);
    }
    __syncthreads();
    for (int idx = tid; idx < 32 * 32; idx += 256) {
      int b = idx >> 5, i = idx & 31;
      float hv = (float)hL[b * 256 + j0 + i];
      hmsh[b * 256 + j0 + i] = (half_t)(2.f * sigf(scr[b * 32 + i]) * hv);
    }
    gen += NWG; gbar(syncCnt, gen);
    for (int i = tid; i < 32 * 256 / 8; i += 256)
      cp16((char*)hmL + i * 16, (const char*)hmsh + i * 16);
    cp_fence();
    __syncthreads();

    // ---- phase 3: gates = x@WihT-slice + h_mog@WhhT-slice (32x128) ----
    {
      int nt = wave;  // each wave: one 16-col strip, both M tiles
      v8f acc0 = {}, acc1 = {};
      for (int k = 0; k < D2P; k += 32) {
        v16h bf = ld_b(wih, D2P, nt * 16, k, lane);
        acc0 = wmma_f16(ld_a(xL, D2P, 0, k, lane), bf, acc0);
        acc1 = wmma_f16(ld_a(xL, D2P, 16, k, lane), bf, acc1);
      }
      for (int k = 0; k < 256; k += 32) {
        v16h bf = ld_b(whh, 256, nt * 16, k, lane);
        acc0 = wmma_f16(ld_a(hmL, 256, 0, k, lane), bf, acc0);
        acc1 = wmma_f16(ld_a(hmL, 256, 16, k, lane), bf, acc1);
      }
      st_d(scr, 128, 0, nt * 16, lane, acc0);
      st_d(scr, 128, 16, nt * 16, lane, acc1);
    }
    __syncthreads();
    for (int idx = tid; idx < 32 * 32; idx += 256) {
      int b = idx >> 5, i = idx & 31;
      float ig = scr[b * 128 + i]       + bL[i];
      float fg = scr[b * 128 + 32 + i]  + bL[32 + i];
      float cg = scr[b * 128 + 64 + i]  + bL[64 + i];
      float og = scr[b * 128 + 96 + i]  + bL[96 + i];
      float c = sigf(fg) * cL[b * 32 + i] + sigf(ig) * tanh_f(cg);
      cL[b * 32 + i] = c;
      half_t hh = (half_t)(sigf(og) * tanh_f(c));
      hsh[b * 256 + j0 + i] = hh;
      hsOut[((size_t)b * 2048 + t) * 256 + j0 + i] = hh;
    }
    gen += NWG; gbar(syncCnt, gen);
    for (int i = tid; i < 32 * 256 / 8; i += 256)
      cp16((char*)hL + i * 16, (const char*)hsh + i * 16);
    cp_fence();
    __syncthreads();
  }
}

// ---------------------------------------------------------------------------
// Kernel 3: conv branches + ReLU + global max. One block per (b, branch, oc).
// max(relu(z+bias)) == relu(max(z)+bias) since ReLU/+bias are monotone.
// ---------------------------------------------------------------------------
__global__ __launch_bounds__(256) void MLog_conv_kernel(
    const half_t* __restrict__ hs, const float* __restrict__ w3,
    const float* __restrict__ b3, const float* __restrict__ w4,
    const float* __restrict__ b4, const float* __restrict__ w5,
    const float* __restrict__ b5, float* __restrict__ feats) {
  __shared__ float wlds[5 * 256];
  __shared__ float red[256];
  int b = blockIdx.x / 9, c9 = blockIdx.x % 9;
  int br = c9 / 3, oc = c9 % 3;
  int f = 3 + br;
  const float* W = (br == 0 ? w3 : (br == 1 ? w4 : w5)) + oc * f * 256;
  float bias = (br == 0 ? b3 : (br == 1 ? b4 : b5))[oc];
  for (int i = threadIdx.x; i < f * 256; i += 256) wlds[i] = W[i];
  __syncthreads();
  float mv = -1e30f;
  int nw = 2048 - f + 1;
  for (int t = threadIdx.x; t < nw; t += 256) {
    const half_t* base = hs + ((size_t)b * 2048 + t) * 256;
    float acc = 0.f;
    for (int dt = 0; dt < f; ++dt) {
      const half_t* hp = base + dt * 256;
      const float* wp = wlds + dt * 256;
#pragma unroll 8
      for (int h = 0; h < 256; ++h) acc += (float)hp[h] * wp[h];
    }
    mv = fmaxf(mv, acc);
  }
  red[threadIdx.x] = mv;
  __syncthreads();
  for (int s = 128; s > 0; s >>= 1) {
    if (threadIdx.x < s)
      red[threadIdx.x] = fmaxf(red[threadIdx.x], red[threadIdx.x + s]);
    __syncthreads();
  }
  if (threadIdx.x == 0) feats[b * 9 + c9] = fmaxf(red[0] + bias, 0.f);
}

// Kernel 4: feats(32x9) @ linW(9x2) + linb
__global__ void MLog_head_kernel(const float* __restrict__ feats,
                                 const float* __restrict__ linW,
                                 const float* __restrict__ linb,
                                 float* __restrict__ out) {
  int tid = threadIdx.x;
  if (tid < 64) {
    int b = tid >> 1, j = tid & 1;
    float s = linb[j];
#pragma unroll
    for (int i = 0; i < 9; ++i) s += feats[b * 9 + i] * linW[i * 2 + j];
    out[b * 2 + j] = s;
  }
}

extern "C" void kernel_launch(void* const* d_in, const int* in_sizes, int n_in,
                              void* d_out, int out_size, void* d_ws,
                              size_t ws_size, hipStream_t stream) {
  (void)in_sizes; (void)n_in; (void)out_size;
  const float* features = (const float*)d_in[0];
  const float* W_in = (const float*)d_in[1];
  const float* b_in = (const float*)d_in[2];
  const float* Q    = (const float*)d_in[3];
  const float* R    = (const float*)d_in[4];
  const float* Wih  = (const float*)d_in[5];
  const float* Whh  = (const float*)d_in[6];
  const float* bih  = (const float*)d_in[7];
  const float* bhh  = (const float*)d_in[8];
  const float* w3 = (const float*)d_in[9],  *b3 = (const float*)d_in[10];
  const float* w4 = (const float*)d_in[11], *b4 = (const float*)d_in[12];
  const float* w5 = (const float*)d_in[13], *b5 = (const float*)d_in[14];
  const float* linW = (const float*)d_in[15];
  const float* linb = (const float*)d_in[16];

  char* ws = (char*)d_ws;
  size_t off = 0;
  auto alloc = [&](size_t bytes) {
    size_t o = off;
    off += (bytes + 255) & ~size_t(255);
    return o;
  };
  size_t oSync = alloc(256);
  size_t oXh   = alloc((size_t)2048 * 32 * D2P * 2);   // projected x, f16
  size_t oQt   = alloc((size_t)D2P * 256 * 2);
  size_t oRt   = alloc((size_t)256 * D2P * 2);
  size_t oWih  = alloc((size_t)1024 * D2P * 2);
  size_t oWhh  = alloc((size_t)1024 * 256 * 2);
  size_t oWin  = alloc((size_t)D2P * KINP * 2);
  size_t oXsh  = alloc(32 * D2P * 2);
  size_t oHm   = alloc(32 * 256 * 2);
  size_t oHsh  = alloc(32 * 256 * 2);
  size_t oHs   = alloc((size_t)32 * 2048 * 256 * 2);   // all hidden states, f16
  size_t oFeat = alloc(32 * 9 * 4);
  if (off > ws_size) return;  // workspace too small; nothing safe to do

  (void)hipMemsetAsync(ws + oSync, 0, 256, stream);  // barrier counter = 0

  // transposed/padded f16 weight images (WMMA-B layout: N-major, K contiguous)
  MLog_prep_kernel<<<192, 256, 0, stream>>>(W_in, (half_t*)(ws + oWin), 300, 300, KINP, D2P);
  MLog_prep_kernel<<<192, 256, 0, stream>>>(Q,    (half_t*)(ws + oQt),  256, 300, 256, D2P);
  MLog_prep_kernel<<<192, 256, 0, stream>>>(R,    (half_t*)(ws + oRt),  300, 256, D2P, 256);
  MLog_prep_kernel<<<768, 256, 0, stream>>>(Wih,  (half_t*)(ws + oWih), 300, 1024, D2P, 1024);
  MLog_prep_kernel<<<512, 256, 0, stream>>>(Whh,  (half_t*)(ws + oWhh), 256, 1024, 256, 1024);

  // input projection (WMMA), 245KB LDS for W_in^T
  MLog_inproj_kernel<<<512, 256, (size_t)D2P * KINP * 2, stream>>>(
      features, b_in, (const half_t*)(ws + oWin), (half_t*)(ws + oXh));

  // persistent Mogrifier-LSTM: 8 WGs, 284.5KB LDS each
  MLog_lstm_kernel<<<NWG, 256, 291328, stream>>>(
      (const half_t*)(ws + oXh), (const half_t*)(ws + oQt),
      (const half_t*)(ws + oRt), (const half_t*)(ws + oWih),
      (const half_t*)(ws + oWhh), bih, bhh, (half_t*)(ws + oXsh),
      (half_t*)(ws + oHm), (half_t*)(ws + oHsh), (half_t*)(ws + oHs),
      (unsigned*)(ws + oSync));

  MLog_conv_kernel<<<288, 256, 0, stream>>>((const half_t*)(ws + oHs), w3, b3,
                                            w4, b4, w5, b5,
                                            (float*)(ws + oFeat));
  MLog_head_kernel<<<1, 64, 0, stream>>>((const float*)(ws + oFeat), linW, linb,
                                         (float*)d_out);
}
